// EncoderLayer_8959301780085
// MI455X (gfx1250) — matrix operand order repaired and checked by reading, not compiled, not run on hardware
//
#include <hip/hip_runtime.h>
#include <hip/hip_bf16.h>
#include <math.h>

// ---------------------------------------------------------------------------
// Problem constants (match reference)
// ---------------------------------------------------------------------------
#define EMBED  1024
#define NHEAD  16
#define HDIM   64
#define FFDIM  4096
#define BATCH  2
#define SEQ    2048
#define MTOT   (BATCH*SEQ)     // 4096 rows
#define LN_EPS 1e-6f

typedef __attribute__((ext_vector_type(16))) __bf16 v16bf;
typedef __attribute__((ext_vector_type(8)))  __bf16 v8bf;
typedef __attribute__((ext_vector_type(8)))  float  v8f;

// ---------------------------------------------------------------------------
// WMMA fragment gathers (wave32 layouts per CDNA5 ISA 7.12.2)
//
// A (16x32 bf16): lane l, half = l>>4, row M = l&15.
//   vector elems [0..7]  = K = half*8 + [0..7]        (16 contiguous bytes)
//   vector elems [8..15] = K = 16 + half*8 + [0..7]   (16 contiguous bytes)
// B (32x16 bf16): lane l, col N = l&15, khalf = l>>4.
//   vector elems [0..7]  = K = khalf*8 + [0..7], [8..15] = K = 16 + khalf*8 + [0..7]
// C/D (16x16 f32): VGPR r, lane l -> row r + 8*(l>>4), col l&15.
// ---------------------------------------------------------------------------
static __device__ __forceinline__ v16bf frag_a(const __bf16* p, int half) {
  v8bf lo = *(const v8bf*)(p + half * 8);
  v8bf hi = *(const v8bf*)(p + 16 + half * 8);
  v16bf r;
#pragma unroll
  for (int i = 0; i < 8; ++i) { r[i] = lo[i]; r[i + 8] = hi[i]; }
  return r;
}

static __device__ __forceinline__ v16bf frag_b(const __bf16* p, int half) {
  v8bf lo = *(const v8bf*)(p + half * 8);
  v8bf hi = *(const v8bf*)(p + 16 + half * 8);
  v16bf r;
#pragma unroll
  for (int i = 0; i < 8; ++i) { r[i] = lo[i]; r[i + 8] = hi[i]; }
  return r;
}

static __device__ __forceinline__ v8f wmma_bf16(v16bf a, v16bf b, v8f c) {
  return __builtin_amdgcn_wmma_f32_16x16x32_bf16(false, a, false, b,
                                                 (short)0, c, false, false);
}

// ---------------------------------------------------------------------------
// fp32 -> bf16 elementwise convert
// ---------------------------------------------------------------------------
__global__ __launch_bounds__(256) void enc_cvt_bf16(
    const float* __restrict__ in, __bf16* __restrict__ out, int n) {
  int i = blockIdx.x * 256 + threadIdx.x;
  if (i < n) out[i] = (__bf16)in[i];
}

// fp32 [K,N] -> bf16 [N,K] (weight transpose so B-fragments are contiguous)
__global__ __launch_bounds__(256) void enc_transpose_bf16(
    const float* __restrict__ W, __bf16* __restrict__ Wt, int K, int N) {
  int i = blockIdx.x * 256 + threadIdx.x;
  if (i < K * N) {
    int k = i / N, n = i % N;
    Wt[(size_t)n * K + k] = (__bf16)W[i];
  }
}

// ---------------------------------------------------------------------------
// Generic bf16 WMMA GEMM:  C[M,N] = act(A[M,K] @ Bt[N,K]^T + bias)
// Block tile 128x128, 8 waves; wave (wm,wn) owns rows [wm*32,+32), cols
// [wn*64,+64) => 2x4 grid of 16x16 WMMA tiles, K-step 32.
// Epilogue: optional exact GELU; writes fp32 row-major, bf16 row-major,
// and/or bf16 head-transposed [B,H,D,S] (for the V matrix).
// ---------------------------------------------------------------------------
__global__ __launch_bounds__(256) void enc_gemm_bf16(
    const __bf16* __restrict__ A, const __bf16* __restrict__ Bt,
    const float* __restrict__ bias,
    float* __restrict__ outF, __bf16* __restrict__ outB,
    __bf16* __restrict__ outVT,
    int M, int N, int K, int do_gelu) {
  const int lane = threadIdx.x & 31;
  const int wave = threadIdx.x >> 5;
  const int half = lane >> 4;
  const int l16  = lane & 15;
  const int wm = wave & 3;
  const int wn = wave >> 2;
  const int row0 = blockIdx.x * 128 + wm * 32;
  const int col0 = blockIdx.y * 128 + wn * 64;

  v8f acc[2][4];
#pragma unroll
  for (int i = 0; i < 2; ++i)
#pragma unroll
    for (int j = 0; j < 4; ++j)
#pragma unroll
      for (int r = 0; r < 8; ++r) acc[i][j][r] = 0.0f;

  const __bf16* arow[2];
  const __bf16* bcol[4];
#pragma unroll
  for (int i = 0; i < 2; ++i)
    arow[i] = A + (size_t)(row0 + i * 16 + l16) * K;
#pragma unroll
  for (int j = 0; j < 4; ++j)
    bcol[j] = Bt + (size_t)(col0 + j * 16 + l16) * K;

  for (int kb = 0; kb < K; kb += 32) {
    v16bf af[2], bf_[4];
#pragma unroll
    for (int i = 0; i < 2; ++i) af[i] = frag_a(arow[i] + kb, half);
#pragma unroll
    for (int j = 0; j < 4; ++j) bf_[j] = frag_b(bcol[j] + kb, half);
#pragma unroll
    for (int i = 0; i < 2; ++i)
#pragma unroll
      for (int j = 0; j < 4; ++j)
        acc[i][j] = wmma_bf16(af[i], bf_[j], acc[i][j]);
  }

#pragma unroll
  for (int i = 0; i < 2; ++i)
#pragma unroll
    for (int j = 0; j < 4; ++j)
#pragma unroll
      for (int r = 0; r < 8; ++r) {
        int rr = row0 + i * 16 + r + 8 * half;
        int cc = col0 + j * 16 + l16;
        float v = acc[i][j][r] + bias[cc];
        if (do_gelu) v = 0.5f * v * (1.0f + erff(v * 0.70710678118654752f));
        if (outF) outF[(size_t)rr * N + cc] = v;
        if (outB) outB[(size_t)rr * N + cc] = (__bf16)v;
        if (outVT) {
          int h = cc >> 6, d = cc & (HDIM - 1);
          int b = rr >> 11, s = rr & (SEQ - 1);
          outVT[((size_t)(b * NHEAD + h) * HDIM + d) * SEQ + s] = (__bf16)v;
        }
      }
}

// ---------------------------------------------------------------------------
// Flash attention, bf16 in / bf16 out, fp32 softmax statistics.
// Q,K: [B,S,H*D] bf16 row-major.  Vt: [B,H,D,S] bf16.  Out: [B,S,H*D] bf16.
// One block = 8 waves; each wave owns 16 queries of one (b,h); streams keys
// in blocks of 32 with online softmax. P-tile goes C-layout -> LDS(bf16) ->
// A-layout reload (ds_store_b16 / ds_load_b128, same-wave in-order DS).
// ---------------------------------------------------------------------------
__global__ __launch_bounds__(256) void enc_flash_attn(
    const __bf16* __restrict__ Qb, const __bf16* __restrict__ Kb,
    const __bf16* __restrict__ Vt, __bf16* __restrict__ Ob) {
  __shared__ __bf16 plds[8 * 16 * 32];   // 8 KB: per-wave 16x32 P tile

  const int lane = threadIdx.x & 31;
  const int wave = threadIdx.x >> 5;
  const int half = lane >> 4;
  const int l16  = lane & 15;

  const int nqb  = SEQ / 128;
  const int qblk = blockIdx.x % nqb;
  const int bh   = blockIdx.x / nqb;
  const int h    = bh % NHEAD;
  const int b    = bh / NHEAD;
  const int q0   = qblk * 128 + wave * 16;

  // Q A-fragments for d-steps {0,32}; reused across the whole key loop.
  const __bf16* qrow =
      Qb + (size_t)(b * SEQ + q0 + l16) * EMBED + h * HDIM;
  v16bf qa0 = frag_a(qrow + 0, half);
  v16bf qa1 = frag_a(qrow + 32, half);

  v8f o[4];
#pragma unroll
  for (int j = 0; j < 4; ++j)
#pragma unroll
    for (int r = 0; r < 8; ++r) o[j][r] = 0.0f;
  float mrow[8], lrow[8];
#pragma unroll
  for (int r = 0; r < 8; ++r) { mrow[r] = -3.0e38f; lrow[r] = 0.0f; }

  __bf16* pbuf = &plds[wave * 512];
  const float scale = 0.125f;  // 1/sqrt(64)

  for (int k0 = 0; k0 < SEQ; k0 += 32) {
    // ---- scores: two 16x16 tiles over 32 keys, K-dim = d (2 steps) ----
    v8f st[2];
#pragma unroll
    for (int t = 0; t < 2; ++t) {
#pragma unroll
      for (int r = 0; r < 8; ++r) st[t][r] = 0.0f;
      const __bf16* krow =
          Kb + (size_t)(b * SEQ + k0 + t * 16 + l16) * EMBED + h * HDIM;
      v16bf kf0 = frag_b(krow + 0, half);
      v16bf kf1 = frag_b(krow + 32, half);
      st[t] = wmma_bf16(qa0, kf0, st[t]);
      st[t] = wmma_bf16(qa1, kf1, st[t]);
    }

    // ---- online softmax update (row reductions across 16-lane halves) ----
    float alpha[8];
#pragma unroll
    for (int r = 0; r < 8; ++r) {
      float s0 = st[0][r] * scale;
      float s1 = st[1][r] * scale;
      float rm = fmaxf(s0, s1);
#pragma unroll
      for (int msk = 1; msk <= 8; msk <<= 1)
        rm = fmaxf(rm, __shfl_xor(rm, msk, 32));
      float mn = fmaxf(mrow[r], rm);
      alpha[r] = __expf(mrow[r] - mn);
      float p0 = __expf(s0 - mn);
      float p1 = __expf(s1 - mn);
      float ps = p0 + p1;
#pragma unroll
      for (int msk = 1; msk <= 8; msk <<= 1)
        ps += __shfl_xor(ps, msk, 32);
      lrow[r] = lrow[r] * alpha[r] + ps;
      mrow[r] = mn;
      // stash P in LDS: row (r + 8*half), key-col (t*16 + l16)
      pbuf[(r + 8 * half) * 32 + l16]      = (__bf16)p0;
      pbuf[(r + 8 * half) * 32 + 16 + l16] = (__bf16)p1;
    }
#pragma unroll
    for (int j = 0; j < 4; ++j)
#pragma unroll
      for (int r = 0; r < 8; ++r) o[j][r] *= alpha[r];

    // same-wave DS RAW: make sure the P stores landed before A-layout reload
    asm volatile("s_wait_dscnt 0" ::: "memory");

    v16bf pa;
    {
      const __bf16* pr = pbuf + l16 * 32;
      v8bf lo = *(const v8bf*)(pr + half * 8);
      v8bf hi = *(const v8bf*)(pr + 16 + half * 8);
#pragma unroll
      for (int i = 0; i < 8; ++i) { pa[i] = lo[i]; pa[i + 8] = hi[i]; }
    }

    // ---- O += P @ V : four 16-wide d tiles, K-dim = 32 keys ----
#pragma unroll
    for (int j = 0; j < 4; ++j) {
      const __bf16* vcol =
          Vt + ((size_t)(b * NHEAD + h) * HDIM + j * 16 + l16) * SEQ + k0;
      v16bf vf = frag_b(vcol, half);
      o[j] = wmma_bf16(pa, vf, o[j]);
    }
  }

  // ---- normalize and emit bf16 [B,S,H*D] ----
#pragma unroll
  for (int j = 0; j < 4; ++j)
#pragma unroll
    for (int r = 0; r < 8; ++r) {
      float val = o[j][r] / lrow[r];
      int row = q0 + r + 8 * half;
      int d   = j * 16 + l16;
      Ob[(size_t)(b * SEQ + row) * EMBED + h * HDIM + d] = (__bf16)val;
    }
}

// ---------------------------------------------------------------------------
// Fused residual-add + LayerNorm, one block (256 thr) per row of 1024.
// Optionally also emits a bf16 copy for the next GEMM's A operand.
// ---------------------------------------------------------------------------
__global__ __launch_bounds__(256) void enc_add_layernorm(
    const float* __restrict__ X, const float* __restrict__ Y,
    const float* __restrict__ gamma, const float* __restrict__ beta,
    float* __restrict__ outF, __bf16* __restrict__ outB) {
  __shared__ float redS[8];
  __shared__ float redQ[8];
  __shared__ float stat[2];

  const int row  = blockIdx.x;
  const int wave = threadIdx.x >> 5;
  const int lane = threadIdx.x & 31;
  const int c0   = threadIdx.x * 4;

  const float* xr = X + (size_t)row * EMBED;
  const float* yr = Y + (size_t)row * EMBED;
  float v[4];
  float s = 0.0f, ss = 0.0f;
#pragma unroll
  for (int i = 0; i < 4; ++i) {
    v[i] = xr[c0 + i] + yr[c0 + i];
    s += v[i];
    ss += v[i] * v[i];
  }
#pragma unroll
  for (int msk = 1; msk < 32; msk <<= 1) {
    s  += __shfl_xor(s, msk, 32);
    ss += __shfl_xor(ss, msk, 32);
  }
  if (lane == 0) { redS[wave] = s; redQ[wave] = ss; }
  __syncthreads();
  if (threadIdx.x == 0) {
    float ts = 0.0f, tq = 0.0f;
#pragma unroll
    for (int w = 0; w < 8; ++w) { ts += redS[w]; tq += redQ[w]; }
    float mu  = ts * (1.0f / EMBED);
    float var = tq * (1.0f / EMBED) - mu * mu;
    stat[0] = mu;
    stat[1] = rsqrtf(var + LN_EPS);
  }
  __syncthreads();
  const float mu = stat[0], rs = stat[1];
#pragma unroll
  for (int i = 0; i < 4; ++i) {
    int c = c0 + i;
    float outv = (v[i] - mu) * rs * gamma[c] + beta[c];
    outF[(size_t)row * EMBED + c] = outv;
    if (outB) outB[(size_t)row * EMBED + c] = (__bf16)outv;
  }
}

// ---------------------------------------------------------------------------
// Host-side orchestration
// ---------------------------------------------------------------------------
extern "C" void kernel_launch(void* const* d_in, const int* in_sizes, int n_in,
                              void* d_out, int out_size, void* d_ws,
                              size_t ws_size, hipStream_t stream) {
  (void)in_sizes; (void)n_in; (void)out_size; (void)ws_size;
  const float* x   = (const float*)d_in[0];
  const float* wq  = (const float*)d_in[1];
  const float* bq  = (const float*)d_in[2];
  const float* wk  = (const float*)d_in[3];
  const float* bk  = (const float*)d_in[4];
  const float* wv  = (const float*)d_in[5];
  const float* bv  = (const float*)d_in[6];
  const float* wo  = (const float*)d_in[7];
  const float* bo  = (const float*)d_in[8];
  const float* gm1 = (const float*)d_in[9];
  const float* bt1 = (const float*)d_in[10];
  const float* w1  = (const float*)d_in[11];
  const float* b1  = (const float*)d_in[12];
  const float* w2  = (const float*)d_in[13];
  const float* b2  = (const float*)d_in[14];
  const float* gm2 = (const float*)d_in[15];
  const float* bt2 = (const float*)d_in[16];

  char* ws = (char*)d_ws;
  const size_t MB = (size_t)1 << 20;
  __bf16* xb    = (__bf16*)(ws + 0 * MB);   //  8 MB  x bf16
  __bf16* wqt   = (__bf16*)(ws + 8 * MB);   //  2 MB
  __bf16* wkt   = (__bf16*)(ws + 10 * MB);  //  2 MB
  __bf16* wvt   = (__bf16*)(ws + 12 * MB);  //  2 MB
  __bf16* wot   = (__bf16*)(ws + 14 * MB);  //  2 MB
  __bf16* w1t   = (__bf16*)(ws + 16 * MB);  //  8 MB
  __bf16* w2t   = (__bf16*)(ws + 24 * MB);  //  8 MB
  __bf16* qb    = (__bf16*)(ws + 32 * MB);  //  8 MB
  __bf16* kb    = (__bf16*)(ws + 40 * MB);  //  8 MB
  __bf16* vt    = (__bf16*)(ws + 48 * MB);  //  8 MB  [B,H,D,S]
  __bf16* attnb = (__bf16*)(ws + 56 * MB);  //  8 MB
  __bf16* tb    = (__bf16*)(ws + 32 * MB);  // 32 MB  aliases qb..attnb (dead)
  float*  attnp = (float*)(ws + 64 * MB);   // 16 MB
  float*  ffp   = (float*)(ws + 64 * MB);   // 16 MB  aliases attnp (dead)
  float*  hF    = (float*)(ws + 80 * MB);   // 16 MB
  __bf16* hb    = (__bf16*)(ws + 96 * MB);  //  8 MB  -> peak 104 MB

  dim3 thr(256);

  // ---- precision staging: fp32 -> bf16 (+ weight transposes to [N,K]) ----
  enc_cvt_bf16<<<(MTOT * EMBED + 255) / 256, thr, 0, stream>>>(x, xb,
                                                               MTOT * EMBED);
  enc_transpose_bf16<<<(EMBED * EMBED + 255) / 256, thr, 0, stream>>>(
      wq, wqt, EMBED, EMBED);
  enc_transpose_bf16<<<(EMBED * EMBED + 255) / 256, thr, 0, stream>>>(
      wk, wkt, EMBED, EMBED);
  enc_transpose_bf16<<<(EMBED * EMBED + 255) / 256, thr, 0, stream>>>(
      wv, wvt, EMBED, EMBED);
  enc_transpose_bf16<<<(EMBED * EMBED + 255) / 256, thr, 0, stream>>>(
      wo, wot, EMBED, EMBED);
  enc_transpose_bf16<<<(EMBED * FFDIM + 255) / 256, thr, 0, stream>>>(
      w1, w1t, EMBED, FFDIM);
  enc_transpose_bf16<<<(FFDIM * EMBED + 255) / 256, thr, 0, stream>>>(
      w2, w2t, FFDIM, EMBED);

  // ---- QKV projections (WMMA) ----
  dim3 gE(MTOT / 128, EMBED / 128);
  enc_gemm_bf16<<<gE, thr, 0, stream>>>(xb, wqt, bq, nullptr, qb, nullptr,
                                        MTOT, EMBED, EMBED, 0);
  enc_gemm_bf16<<<gE, thr, 0, stream>>>(xb, wkt, bk, nullptr, kb, nullptr,
                                        MTOT, EMBED, EMBED, 0);
  enc_gemm_bf16<<<gE, thr, 0, stream>>>(xb, wvt, bv, nullptr, nullptr, vt,
                                        MTOT, EMBED, EMBED, 0);

  // ---- flash attention (WMMA QK^T and PV) ----
  enc_flash_attn<<<BATCH * NHEAD * (SEQ / 128), thr, 0, stream>>>(qb, kb, vt,
                                                                  attnb);

  // ---- output projection + residual/LN 1 ----
  enc_gemm_bf16<<<gE, thr, 0, stream>>>(attnb, wot, bo, attnp, nullptr,
                                        nullptr, MTOT, EMBED, EMBED, 0);
  enc_add_layernorm<<<MTOT, thr, 0, stream>>>(x, attnp, gm1, bt1, hF, hb);

  // ---- FFN: gelu(h@W1+b1)@W2+b2, then residual/LN 2 into d_out ----
  dim3 gF(MTOT / 128, FFDIM / 128);
  enc_gemm_bf16<<<gF, thr, 0, stream>>>(hb, w1t, b1, nullptr, tb, nullptr,
                                        MTOT, FFDIM, EMBED, 1);
  enc_gemm_bf16<<<gE, thr, 0, stream>>>(tb, w2t, b2, ffp, nullptr, nullptr,
                                        MTOT, EMBED, FFDIM, 0);
  enc_add_layernorm<<<MTOT, thr, 0, stream>>>(hF, ffp, gm2, bt2,
                                              (float*)d_out, nullptr);
}
